// CTroid_55843164783484
// MI455X (gfx1250) — matrix operand order repaired and checked by reading, not compiled, not run on hardware
//
#include <hip/hip_runtime.h>

// ---------------------------------------------------------------------------
// Types
// ---------------------------------------------------------------------------
typedef __attribute__((ext_vector_type(16))) __bf16 v16bf;
typedef __attribute__((ext_vector_type(8)))  float  v8f;

union FragBF16 {
    uint4 q[2];
    v16bf v;
};

__device__ __forceinline__ unsigned short f32_to_bf16(float x) {
    unsigned int u = __float_as_uint(x);
    unsigned int r = (u + 0x7FFFu + ((u >> 16) & 1u)) >> 16;  // RNE
    return (unsigned short)r;
}
__device__ __forceinline__ float bf16_to_f32(unsigned short h) {
    return __uint_as_float(((unsigned int)h) << 16);
}

// CDNA5 async copy (GVS mode): 2 x 16 B global -> LDS, ASYNCcnt-tracked.
// INST_OFFSET is added to BOTH the global and the LDS address (ISA 08 §4.4),
// so one address pair serves a 32 B row segment via offset:0 / offset:16.
__device__ __forceinline__ void async_pair(unsigned lds_off, unsigned voff,
                                           const unsigned short* sbase) {
    asm volatile("global_load_async_to_lds_b128 %0, %1, %2\n\t"
                 "global_load_async_to_lds_b128 %0, %1, %2 offset:16"
                 :: "v"(lds_off), "v"(voff), "s"(sbase) : "memory");
}
__device__ __forceinline__ void wait_async0() {
    asm volatile("s_wait_asynccnt 0x0" ::: "memory");
}

// ---------------------------------------------------------------------------
// Kernel 1: split fp32 matrix rows into bf16 hi/lo + row sum-of-squares.
// One wave32 per row; rows in [rows_src, rows_pad) are zero-filled padding.
// ---------------------------------------------------------------------------
__global__ __launch_bounds__(256) void prep_split(
    const float* __restrict__ src, int rows_src, int rows_pad, int cols,
    unsigned short* __restrict__ hi, unsigned short* __restrict__ lo,
    float* __restrict__ sq)
{
    const int wave = threadIdx.x >> 5;
    const int lane = threadIdx.x & 31;
    const int r = blockIdx.x * 8 + wave;
    if (r >= rows_pad) return;

    const size_t base = (size_t)r * cols;
    float s = 0.0f;
    if (r < rows_src) {
        for (int i = lane; i < cols; i += 32) {
            float x = src[base + i];
            unsigned short h = f32_to_bf16(x);
            unsigned short l = f32_to_bf16(x - bf16_to_f32(h));
            hi[base + i] = h;
            lo[base + i] = l;
            s += x * x;
        }
    } else {
        for (int i = lane; i < cols; i += 32) {
            hi[base + i] = 0;
            lo[base + i] = 0;
        }
    }
    for (int off = 16; off > 0; off >>= 1)
        s += __shfl_down(s, off, 32);
    if (lane == 0) sq[r] = s;
}

// ---------------------------------------------------------------------------
// Kernel 2: tiled bf16-split WMMA GEMM, double-buffered async-LDS pipeline.
//   cross = D @ W^T  via  hi*hi + hi*lo + lo*hi   (fp32 accumulate)
//   out[b,k] = (2*cross - dsq[b] - wsq[k]) * gamma[k]
// Block: 256 threads (8 waves). Tile: 128(M) x 128(N). K-chunk: 32.
// ---------------------------------------------------------------------------
#define F_DIM   512
#define KC      32
#define NCHUNKS (F_DIM / KC)
#define LDSS    40          // padded row stride in bf16 elements (80 B)
#define TILE_Q  (128 * 5)   // one 128x40-elem tile in uint4 units (10 KB)

__global__ __launch_bounds__(256) void gemm_wmma(
    const unsigned short* __restrict__ Ahi, const unsigned short* __restrict__ Alo,
    const unsigned short* __restrict__ Bhi, const unsigned short* __restrict__ Blo,
    const float* __restrict__ d_sq, const float* __restrict__ w_sq,
    const float* __restrict__ gamma, float* __restrict__ out, int Kout)
{
    // double buffer x 4 tiles (Ahi, Alo, Bhi, Blo) = 2 x 40 KB = 80 KB LDS
    __shared__ uint4 sbuf4[2][4 * TILE_Q];

    const int t      = threadIdx.x;
    const int wave   = t >> 5;
    const int lane   = t & 31;
    const int laneHi = lane >> 4;   // 0: lanes 0-15, 1: lanes 16-31
    const int lrow   = lane & 15;

    const int m_base = blockIdx.y * 128;
    const int n_base = blockIdx.x * 128;

    // cooperative-copy coordinates: 2 threads per tile row, 32 B each
    const int crow  = t >> 1;
    const int chalf = t & 1;
    // byte offsets from the matrix bases (add 2*f0 bytes per chunk)
    const unsigned gbA = (unsigned)(((m_base + crow) * F_DIM + chalf * 16) * 2);
    const unsigned gbB = (unsigned)(((n_base + crow) * F_DIM + chalf * 16) * 2);
    const unsigned ldsTileOff = (unsigned)(crow * 5 + chalf * 2) * 16; // bytes

    v8f acc[8];
#pragma unroll
    for (int i = 0; i < 8; ++i) acc[i] = (v8f)(0.0f);

    // issue one chunk's 4 tiles (2 x 16 B per tile per thread) into buffer b
    auto issue_chunk = [&](int f0, int b) {
        const unsigned fb = (unsigned)(f0 * 2);  // byte offset of this chunk
        unsigned base = (unsigned)(uintptr_t)&sbuf4[b][0];
        async_pair(base + 0 * TILE_Q * 16 + ldsTileOff, gbA + fb, Ahi);
        async_pair(base + 1 * TILE_Q * 16 + ldsTileOff, gbA + fb, Alo);
        async_pair(base + 2 * TILE_Q * 16 + ldsTileOff, gbB + fb, Bhi);
        async_pair(base + 3 * TILE_Q * 16 + ldsTileOff, gbB + fb, Blo);
    };

    // prologue: chunk 0 -> buffer 0
    issue_chunk(0, 0);

    int cur = 0;
    for (int ci = 0; ci < NCHUNKS; ++ci) {
        wait_async0();        // my async copies for chunk ci have landed
        __syncthreads();      // everyone's copies landed; prev compute done
        if (ci + 1 < NCHUNKS) issue_chunk((ci + 1) * KC, cur ^ 1);

        const unsigned short* sAhi = (const unsigned short*)&sbuf4[cur][0 * TILE_Q];
        const unsigned short* sAlo = (const unsigned short*)&sbuf4[cur][1 * TILE_Q];
        const unsigned short* sBhi = (const unsigned short*)&sbuf4[cur][2 * TILE_Q];
        const unsigned short* sBlo = (const unsigned short*)&sbuf4[cur][3 * TILE_Q];

        // ---- A fragments (16x32 bf16, CDNA5 layout) for this wave's M-strip
        // lane<16 : V0-3 = K0..7   (bytes 0..15),  V4-7 = K16..23 (bytes 32..47)
        // lane>=16: V0-3 = K8..15  (bytes 16..31), V4-7 = K24..31 (bytes 48..63)
        FragBF16 fah, fal;
        {
            const unsigned short* arh = sAhi + (wave * 16 + lrow) * LDSS;
            const unsigned short* arl = sAlo + (wave * 16 + lrow) * LDSS;
            fah.q[0] = *(const uint4*)(arh + laneHi * 8);
            fah.q[1] = *(const uint4*)(arh + 16 + laneHi * 8);
            fal.q[0] = *(const uint4*)(arl + laneHi * 8);
            fal.q[1] = *(const uint4*)(arl + 16 + laneHi * 8);
        }

#pragma unroll
        for (int n = 0; n < 8; ++n) {
            // ---- B fragment (32x16 bf16): lane<16 = K0..7,K16..23 of col lrow,
            //      lane>=16 = K8..15,K24..31 (same order as A)
            FragBF16 fbh, fbl;
            const unsigned short* brh = sBhi + (n * 16 + lrow) * LDSS;
            const unsigned short* brl = sBlo + (n * 16 + lrow) * LDSS;
            fbh.q[0] = *(const uint4*)(brh + laneHi * 8);
            fbh.q[1] = *(const uint4*)(brh + 16 + laneHi * 8);
            fbl.q[0] = *(const uint4*)(brl + laneHi * 8);
            fbl.q[1] = *(const uint4*)(brl + 16 + laneHi * 8);

            acc[n] = __builtin_amdgcn_wmma_f32_16x16x32_bf16(
                false, fah.v, false, fbh.v, (short)0, acc[n], false, false);
            acc[n] = __builtin_amdgcn_wmma_f32_16x16x32_bf16(
                false, fah.v, false, fbl.v, (short)0, acc[n], false, false);
            acc[n] = __builtin_amdgcn_wmma_f32_16x16x32_bf16(
                false, fal.v, false, fbh.v, (short)0, acc[n], false, false);
        }
        cur ^= 1;
    }

    // ---- fused epilogue: out = (2*cross - dsq - wsq) * gamma ----
    // C/D layout: VGPR v -> M = v + 8*laneHi, N = lrow
    float dsq[8];
#pragma unroll
    for (int v = 0; v < 8; ++v) {
        int m = m_base + wave * 16 + laneHi * 8 + v;
        dsq[v] = d_sq[m];
    }
#pragma unroll
    for (int n = 0; n < 8; ++n) {
        int k = n_base + n * 16 + lrow;
        bool valid = (k < Kout);
        float wsq = w_sq[k];                 // padded to 1024 rows: always safe
        float g   = valid ? gamma[k] : 0.0f; // gamma only has Kout entries
#pragma unroll
        for (int v = 0; v < 8; ++v) {
            int m = m_base + wave * 16 + laneHi * 8 + v;
            float val = (2.0f * acc[n][v] - dsq[v] - wsq) * g;
            if (valid) out[(size_t)m * Kout + k] = val;
        }
    }
}

// ---------------------------------------------------------------------------
// Host launcher
// ---------------------------------------------------------------------------
extern "C" void kernel_launch(void* const* d_in, const int* in_sizes, int n_in,
                              void* d_out, int out_size, void* d_ws, size_t ws_size,
                              hipStream_t stream) {
    (void)in_sizes; (void)n_in; (void)out_size; (void)ws_size;

    const int B = 16384, F = 512, K = 1000, Kpad = 1024;

    const float* D     = (const float*)d_in[0];
    const float* W     = (const float*)d_in[1];
    const float* gamma = (const float*)d_in[2];
    float*       out   = (float*)d_out;

    // workspace layout (~34.1 MB total)
    unsigned short* Dhi = (unsigned short*)d_ws;
    unsigned short* Dlo = Dhi + (size_t)B * F;
    unsigned short* Whi = Dlo + (size_t)B * F;
    unsigned short* Wlo = Whi + (size_t)Kpad * F;
    float*          dsq = (float*)(Wlo + (size_t)Kpad * F);
    float*          wsq = dsq + B;

    prep_split<<<B / 8,    256, 0, stream>>>(D, B, B,    F, Dhi, Dlo, dsq);
    prep_split<<<Kpad / 8, 256, 0, stream>>>(W, K, Kpad, F, Whi, Wlo, wsq);

    dim3 grid(Kpad / 128, B / 128);  // (8, 128)
    gemm_wmma<<<grid, 256, 0, stream>>>(Dhi, Dlo, Whi, Wlo, dsq, wsq, gamma, out, K);
}
